// SelfAttention_33019708571866
// MI455X (gfx1250) — matrix operand order repaired and checked by reading, not compiled, not run on hardware
//
#include <hip/hip_runtime.h>
#include <hip/hip_bf16.h>

// Problem constants (match reference)
constexpr int B = 4, S = 2048, D = 1024, H = 8, E = 128;

typedef __bf16 bf16_t;
typedef bf16_t v16bf __attribute__((ext_vector_type(16)));
typedef bf16_t v8bfv __attribute__((ext_vector_type(8)));
typedef float  v8f   __attribute__((ext_vector_type(8)));
typedef int    v4i   __attribute__((vector_size(16)));

union ABFrag { v16bf v; v8bfv h[2]; };

// LDS row pitches (elements) with +8 bf16 (16B) pad to break bank conflicts
constexpr int LDK = 128 + 8;  // K staging: 32 rows x 128 (ld E)
constexpr int LDV = 32 + 8;   // V^T / W^T staging: 128 rows x 32

// ---- async global->LDS path (gfx1250), with synchronous fallback ----
#if defined(__AMDGCN__) && __has_builtin(__builtin_amdgcn_global_load_async_to_lds_b128)
#define HAVE_ASYNC 1
__device__ inline void g2l_b128(const bf16_t* g, bf16_t* l) {
  __builtin_amdgcn_global_load_async_to_lds_b128(
      (__attribute__((address_space(1))) v4i*)g,
      (__attribute__((address_space(3))) v4i*)l, 0, 0);
}
#define WAIT_ASYNC_4() asm volatile("s_wait_asynccnt 0x4" ::: "memory")
#define WAIT_ASYNC_2() asm volatile("s_wait_asynccnt 0x2" ::: "memory")
#define WAIT_ASYNC_0() asm volatile("s_wait_asynccnt 0x0" ::: "memory")
#else
#define HAVE_ASYNC 0
__device__ inline void g2l_b128(const bf16_t* g, bf16_t* l) {
  *(v8bfv*)l = *(const v8bfv*)g;  // global_load_b128 + ds_store_b128
}
#define WAIT_ASYNC_4()
#define WAIT_ASYNC_2()
#define WAIT_ASYNC_0()
#endif

__device__ inline v8f zero8() {
  v8f z;
#pragma unroll
  for (int i = 0; i < 8; ++i) z[i] = 0.0f;
  return z;
}

// A-matrix fragment: 16x32 bf16, rows M = lane&15, 16 contiguous K at kbase+16*half.
__device__ inline v16bf load_a_bf16(const bf16_t* base, int ld, int kbase, int lane) {
  const int row = lane & 15, hlf = lane >> 4;
  const bf16_t* p = base + (size_t)row * ld + kbase + hlf * 16;
  ABFrag f;
  f.h[0] = *(const v8bfv*)(p);
  f.h[1] = *(const v8bfv*)(p + 8);
  return f.v;
}

// B-matrix fragment: 32x16 bf16, lane holds column N = lane&15, 16 contiguous K at kbase+16*half.
// Memory layout: mem[n][k] row-major (K contiguous per column), ld = elements per n-row.
__device__ inline v16bf load_b_bf16(const bf16_t* base, int ld, int kbase, int lane) {
  const int col = lane & 15, hlf = lane >> 4;
  const bf16_t* p = base + (size_t)col * ld + kbase + hlf * 16;
  ABFrag f;
  f.h[0] = *(const v8bfv*)(p);
  f.h[1] = *(const v8bfv*)(p + 8);
  return f.v;
}

__device__ inline v8f wmma_bf16(v16bf a, v16bf b, v8f c) {
  return __builtin_amdgcn_wmma_f32_16x16x32_bf16(false, a, false, b, (short)0, c, false, false);
}

// ---------------- conversion kernels ----------------
__global__ void cvt_x_kernel(const float* __restrict__ x, bf16_t* __restrict__ xbf, int n) {
  int i = blockIdx.x * blockDim.x + threadIdx.x;
  if (i < n) xbf[i] = (bf16_t)x[i];
}

// W: [H, D, E] f32  ->  WT: [H, E, D] bf16
__global__ void cvt_wT_kernel(const float* __restrict__ W, bf16_t* __restrict__ WT) {
  int i = blockIdx.x * blockDim.x + threadIdx.x;
  if (i >= H * D * E) return;
  int e = i % E;
  int d = (i / E) % D;
  int h = i / (E * D);
  WT[((size_t)h * E + e) * D + d] = (bf16_t)W[i];
}

// ---------------- QKV projection ----------------
// out = x @ W (+bias). Block: 128 rows of S; wave: 16 rows x full E=128.
// Weight chunk (128 e-rows x 32 d) is staged once per block into LDS, double-buffered;
// x A-fragments are software-pipelined one iteration ahead.
__global__ __launch_bounds__(256) void proj_kernel(const bf16_t* __restrict__ xbf,
                                                   const bf16_t* __restrict__ WT,
                                                   const float* __restrict__ bias,
                                                   bf16_t* __restrict__ out, int mode) {
  __shared__ bf16_t wls[2][128 * LDV];

  const int tid = threadIdx.x;
  const int lane = tid & 31;
  const int wv = tid >> 5;
  int bid = blockIdx.x;
  const int ST = S / 128;
  const int st = bid % ST; bid /= ST;
  const int h = bid % H;
  const int b = bid / H;
  const int s0 = st * 128 + wv * 16;

  const bf16_t* xrow = xbf + (size_t)(b * S + s0) * D;
  const bf16_t* wbase = WT + (size_t)h * E * D;

  // stage weight chunk [0..128) x [dd, dd+32): 512 x b128, exactly 2 per thread (no predication)
  const int we0 = tid >> 2, wseg0 = tid & 3;          // j = tid
  const int we1 = (tid + 256) >> 2;                   // j = tid + 256 (same seg)
  auto stage_w = [&](int buf, int dd) {
    g2l_b128(wbase + (size_t)we0 * D + dd + wseg0 * 8, &wls[buf][we0 * LDV + wseg0 * 8]);
    g2l_b128(wbase + (size_t)we1 * D + dd + wseg0 * 8, &wls[buf][we1 * LDV + wseg0 * 8]);
  };

  v8f acc[8];
#pragma unroll
  for (int f = 0; f < 8; ++f) acc[f] = zero8();

  stage_w(0, 0);
  v16bf a_cur = load_a_bf16(xrow, D, 0, lane);
  for (int dd = 0; dd < D; dd += 32) {
    const int cur = (dd >> 5) & 1;
    const bool more = (dd + 32 < D);
    if (more) {
      stage_w(cur ^ 1, dd + 32);
      WAIT_ASYNC_2();
    } else {
      WAIT_ASYNC_0();
    }
    __syncthreads();

    // issue next x fragment early; latency hides under this chunk's WMMAs
    v16bf a_next = a_cur;
    if (more) a_next = load_a_bf16(xrow, D, dd + 32, lane);

    const bf16_t* wc = &wls[cur][0];
    v16bf bw[8];
#pragma unroll
    for (int f = 0; f < 8; ++f) bw[f] = load_b_bf16(wc + f * 16 * LDV, LDV, 0, lane);
#pragma unroll
    for (int f = 0; f < 8; ++f) acc[f] = wmma_bf16(a_cur, bw[f], acc[f]);

    a_cur = a_next;
    __syncthreads();
  }

  const int col = lane & 15, hlf = lane >> 4;
#pragma unroll
  for (int f = 0; f < 8; ++f) {
    const int e = f * 16 + col;
    const float bv = bias ? bias[h * E + e] : 0.0f;
#pragma unroll
    for (int r = 0; r < 8; ++r) {
      const int srow = s0 + r + 8 * hlf;
      const float val = acc[f][r] + bv;
      if (mode == 2)
        out[((size_t)(b * H + h) * E + e) * S + srow] = (bf16_t)val;
      else
        out[((size_t)(b * H + h) * S + srow) * E + e] = (bf16_t)val;
    }
  }
}

// ---------------- flash attention ----------------
// Block: one (b,h), 128 query rows (8 waves x 16). K and V^T chunks (32 keys) are staged
// into LDS once per block (async, double-buffered) and shared by all 8 waves.
// Wave computes S^T = K*Q^T (row stats per-lane + one shfl_xor(16)); P^T is repacked
// in-registers into the B fragment of z^T = V^T * P^T with online-softmax rescaling.
__global__ __launch_bounds__(256) void attn_kernel(const bf16_t* __restrict__ q,
                                                   const bf16_t* __restrict__ k,
                                                   const bf16_t* __restrict__ vT,
                                                   float* __restrict__ out) {
  __shared__ bf16_t kls[2][32 * LDK];
  __shared__ bf16_t vls[2][128 * LDV];

  const int tid = threadIdx.x;
  const int lane = tid & 31;
  const int wv = tid >> 5;
  int bid = blockIdx.x;
  const int ST = S / 128;
  const int st = bid % ST; bid /= ST;
  const int h = bid % H;
  const int b = bid / H;
  const int s0 = st * 128 + wv * 16;
  const int hlf = lane >> 4;

  const size_t bh = (size_t)(b * H + h);
  const bf16_t* qb = q + (bh * S + s0) * E;
  const bf16_t* kgl = k + bh * S * E;
  const bf16_t* vgl = vT + bh * (size_t)E * S;

  // K chunk: 32 rows x 256B -> 512 x b128 (2/thread); V^T chunk: 128 rows x 64B -> 512 x b128 (2/thread)
  const int krow0 = tid >> 4, kseg0 = tid & 15;
  const int krow1 = (tid + 256) >> 4;
  const int ve0 = tid >> 2, vseg0 = tid & 3;
  const int ve1 = (tid + 256) >> 2;
  auto stage = [&](int buf, int t0) {
    g2l_b128(kgl + (size_t)(t0 + krow0) * E + kseg0 * 8, &kls[buf][krow0 * LDK + kseg0 * 8]);
    g2l_b128(kgl + (size_t)(t0 + krow1) * E + kseg0 * 8, &kls[buf][krow1 * LDK + kseg0 * 8]);
    g2l_b128(vgl + (size_t)ve0 * S + t0 + vseg0 * 8, &vls[buf][ve0 * LDV + vseg0 * 8]);
    g2l_b128(vgl + (size_t)ve1 * S + t0 + vseg0 * 8, &vls[buf][ve1 * LDV + vseg0 * 8]);
  };

  // Q^T held as B fragments (lane column = query row s0 + (lane&15))
  v16bf bq[4];
#pragma unroll
  for (int kk = 0; kk < 4; ++kk) bq[kk] = load_b_bf16(qb, E, kk * 32, lane);

  v8f z[8];
#pragma unroll
  for (int f = 0; f < 8; ++f) z[f] = zero8();
  float m = -1e30f, l = 0.0f;
  // 1/sqrt(E) * log2(e): softmax in exp2 domain
  const float sc = 0.08838834764831845f * 1.4426950408889634f;

  stage(0, 0);
  for (int t0 = 0; t0 < S; t0 += 32) {
    const int cur = (t0 >> 5) & 1;
    if (t0 + 32 < S) {
      stage(cur ^ 1, t0 + 32);
      WAIT_ASYNC_4();
    } else {
      WAIT_ASYNC_0();
    }
    __syncthreads();

    const bf16_t* kc = &kls[cur][0];
    const bf16_t* vc = &vls[cur][0];

    // load all 8 K fragments, then run 8 back-to-back WMMAs
    v16bf ak[8];
#pragma unroll
    for (int kk = 0; kk < 4; ++kk) {
      ak[kk]     = load_a_bf16(kc, LDK, kk * 32, lane);
      ak[4 + kk] = load_a_bf16(kc + 16 * LDK, LDK, kk * 32, lane);
    }
    v8f c0 = zero8(), c1 = zero8();
#pragma unroll
    for (int kk = 0; kk < 4; ++kk) {
      c0 = wmma_bf16(ak[kk], bq[kk], c0);
      c1 = wmma_bf16(ak[4 + kk], bq[kk], c1);
    }

    float mx = -1e30f;
#pragma unroll
    for (int r = 0; r < 8; ++r) {
      c0[r] *= sc; c1[r] *= sc;
      mx = fmaxf(mx, fmaxf(c0[r], c1[r]));
    }
    mx = fmaxf(mx, __shfl_xor(mx, 16, 32));
    const float mn = fmaxf(m, mx);
    const float alpha = exp2f(m - mn);
    m = mn;

    float p0[8], p1[8];
    float rs = 0.0f;
#pragma unroll
    for (int r = 0; r < 8; ++r) {
      p0[r] = exp2f(c0[r] - mn);
      p1[r] = exp2f(c1[r] - mn);
      rs += p0[r] + p1[r];
    }
    rs += __shfl_xor(rs, 16, 32);
    l = l * alpha + rs;

#pragma unroll
    for (int f = 0; f < 8; ++f)
#pragma unroll
      for (int r = 0; r < 8; ++r) z[f][r] *= alpha;

    // Repack P^T (C layout) into the B fragment of the PV matmul (single cross-half shuffle).
    ABFrag pb;
#pragma unroll
    for (int r = 0; r < 8; ++r) {
      const float o0 = __shfl_xor(p0[r], 16, 32);
      const float o1 = __shfl_xor(p1[r], 16, 32);
      const float lo = hlf ? o1 : p0[r];   // K local r   (rows 0..7 of this half's tile)
      const float hi = hlf ? p1[r] : o0;   // K local 8+r (rows 8..15)
      pb.v[r] = (bf16_t)lo;
      pb.v[8 + r] = (bf16_t)hi;
    }

    // load all 8 V^T fragments, then run 8 back-to-back WMMAs
    v16bf av[8];
#pragma unroll
    for (int f = 0; f < 8; ++f) av[f] = load_a_bf16(vc + f * 16 * LDV, LDV, 0, lane);
#pragma unroll
    for (int f = 0; f < 8; ++f) z[f] = wmma_bf16(av[f], pb.v, z[f]);

    __syncthreads();
  }

  const float inv = 1.0f / l;
  const int s = s0 + (lane & 15);
#pragma unroll
  for (int f = 0; f < 8; ++f)
#pragma unroll
    for (int r = 0; r < 8; ++r) {
      const int e = f * 16 + r + 8 * hlf;
      out[((size_t)b * S + s) * (H * E) + h * E + e] = z[f][r] * inv;
    }
}

extern "C" void kernel_launch(void* const* d_in, const int* in_sizes, int n_in,
                              void* d_out, int out_size, void* d_ws, size_t ws_size,
                              hipStream_t stream) {
  (void)in_sizes; (void)n_in; (void)out_size; (void)ws_size;
  const float* x  = (const float*)d_in[0];
  const float* Wq = (const float*)d_in[1];
  const float* Wk = (const float*)d_in[2];
  const float* bk = (const float*)d_in[3];
  const float* Wv = (const float*)d_in[4];
  const float* bv = (const float*)d_in[5];
  float* out = (float*)d_out;

  char* ws = (char*)d_ws;
  size_t off = 0;
  bf16_t* xbf = (bf16_t*)(ws + off); off += (size_t)B * S * D * 2;
  bf16_t* WqT = (bf16_t*)(ws + off); off += (size_t)H * E * D * 2;
  bf16_t* WkT = (bf16_t*)(ws + off); off += (size_t)H * E * D * 2;
  bf16_t* WvT = (bf16_t*)(ws + off); off += (size_t)H * E * D * 2;
  bf16_t* qb  = (bf16_t*)(ws + off); off += (size_t)B * H * S * E * 2;
  bf16_t* kb  = (bf16_t*)(ws + off); off += (size_t)B * H * S * E * 2;
  bf16_t* vb  = (bf16_t*)(ws + off); off += (size_t)B * H * S * E * 2;

  const int nx = B * S * D;
  cvt_x_kernel<<<(nx + 255) / 256, 256, 0, stream>>>(x, xbf, nx);
  const int nw = H * D * E;
  cvt_wT_kernel<<<(nw + 255) / 256, 256, 0, stream>>>(Wq, WqT);
  cvt_wT_kernel<<<(nw + 255) / 256, 256, 0, stream>>>(Wk, WkT);
  cvt_wT_kernel<<<(nw + 255) / 256, 256, 0, stream>>>(Wv, WvT);

  const int blocks = B * H * (S / 128);
  proj_kernel<<<blocks, 256, 0, stream>>>(xbf, WqT, nullptr, qb, 0);
  proj_kernel<<<blocks, 256, 0, stream>>>(xbf, WkT, bk, kb, 0);
  proj_kernel<<<blocks, 256, 0, stream>>>(xbf, WvT, bv, vb, 2);

  attn_kernel<<<blocks, 256, 0, stream>>>(qb, kb, vb, out);
}
